// DilatedAttention_13864154431563
// MI455X (gfx1250) — matrix operand order repaired and checked by reading, not compiled, not run on hardware
//
#include <hip/hip_runtime.h>
#include <hip/hip_bf16.h>

typedef __attribute__((ext_vector_type(16))) _Float16 v16h;
typedef __attribute__((ext_vector_type(8)))  _Float16 v8h;
typedef __attribute__((ext_vector_type(8)))  float    v8f;

#define DEV static __device__ __forceinline__

// ---------- WMMA helper (CDNA5 v_wmma_f32_16x16x32_f16) ----------
DEV v8f wmma_f16(v16h a, v16h b, v8f c) {
    return __builtin_amdgcn_wmma_f32_16x16x32_f16(
        /*neg_a=*/false, a, /*neg_b=*/false, b,
        /*c_mod=*/(short)0, c, /*reuse_a=*/false, /*reuse_b=*/false);
}

// A-matrix fragment (16x32 f16): lane = row (lane&15); the lane
// holds K = {c0..c0+15} with c0 = (lane>=16 ? 16 : 0).
DEV v16h load_fragA(const _Float16* row, int lane) {
    int c0 = (lane & 16) ? 16 : 0;
    return *(const v16h*)(row + c0);
}

// B-matrix fragment (32x16 f16): lane = column (lane&15); lanes 0-15 hold
// K=0..15, lanes 16-31 hold K=16..31 (contiguous 16 halves per lane).
DEV v16h load_fragB(const _Float16* span, int lane) {
    int c0 = (lane & 16) ? 16 : 0;
    return *(const v16h*)(span + c0);
}

// XOR-butterfly max over the 16 lanes of a half-wave via ds_swizzle
// (offset = xor<<10 | and 0x1F : lane reads lane^xor).
DEV float redmax16(float x) {
    x = fmaxf(x, __uint_as_float(__builtin_amdgcn_ds_swizzle(__float_as_uint(x), 0x041F)));
    x = fmaxf(x, __uint_as_float(__builtin_amdgcn_ds_swizzle(__float_as_uint(x), 0x081F)));
    x = fmaxf(x, __uint_as_float(__builtin_amdgcn_ds_swizzle(__float_as_uint(x), 0x101F)));
    x = fmaxf(x, __uint_as_float(__builtin_amdgcn_ds_swizzle(__float_as_uint(x), 0x201F)));
    return x;
}

// ---------- constants ----------
#define LSEQ 4096
#define NH   16
#define HD   64

// ---------- xPos rotary prep for Q and K (fp32 -> f16 [h][p][64]) ----------
__global__ __launch_bounds__(256)
void dilattn_prep_qk(const float* __restrict__ q, const float* __restrict__ k,
                     _Float16* __restrict__ q16, _Float16* __restrict__ k16) {
    int idx = blockIdx.x * blockDim.x + threadIdx.x;   // NH*LSEQ*32 pairs
    if (idx >= NH * LSEQ * 32) return;
    int i = idx & 31;               // dim-pair index (0..31)
    int p = (idx >> 5) & (LSEQ - 1);
    int h = idx >> 17;

    float sv    = (2.0f * i + 0.4f * 64.0f) / (1.4f * 64.0f);
    float power = ((float)p - 2048.0f) * (1.0f / 512.0f);
    float scale = __powf(sv, power);
    float invf  = __powf(10000.0f, -(float)i / 32.0f);
    float ang   = (float)p * invf;
    float sn, cs;
    __sincosf(ang, &sn, &cs);

    long src = (long)p * (NH * HD) + h * HD + 2 * i;
    long dst = ((long)h * LSEQ + p) * HD + 2 * i;

    float q1 = q[src], q2 = q[src + 1];
    float cq = cs * scale, sq = sn * scale;
    // fold scaling = d^-0.5 = 0.125 into q
    q16[dst]     = (_Float16)((q1 * cq - q2 * sq) * 0.125f);
    q16[dst + 1] = (_Float16)((q2 * cq + q1 * sq) * 0.125f);

    float k1 = k[src], k2 = k[src + 1];
    float is = 1.0f / scale;
    float ck = cs * is, sk = sn * is;
    k16[dst]     = (_Float16)(k1 * ck - k2 * sk);
    k16[dst + 1] = (_Float16)(k2 * ck + k1 * sk);
}

// ---------- gathered + transposed V per branch: vT[h][d][sg] ----------
__global__ __launch_bounds__(256)
void dilattn_prep_v(const float* __restrict__ v, _Float16* __restrict__ vT,
                    int r, int sl, int nseg, int cshift) {
    int SeqB = nseg * 1024;
    int idx = blockIdx.x * blockDim.x + threadIdx.x;   // NH*HD*SeqB
    if (idx >= NH * HD * SeqB) return;
    int sg = idx % SeqB;
    int d  = (idx / SeqB) & (HD - 1);
    int h  = idx / (SeqB * HD);
    int seg = sg >> 10;
    int s   = sg & 1023;
    int c   = h >> cshift;
    long p  = (long)seg * sl + (long)s * r + c;
    vT[((long)(h * HD + d)) * SeqB + sg] =
        (_Float16)(v[p * (NH * HD) + h * HD + d]);
}

// ---------- flash attention over one branch (causal, dilated gather) ----------
__global__ __launch_bounds__(256)
void dilattn_attn(const _Float16* __restrict__ q16, const _Float16* __restrict__ k16,
                  const _Float16* __restrict__ vT, float* __restrict__ outc,
                  float* __restrict__ lsec, int r, int sl, int nseg, int cshift) {
    const int SeqB = nseg * 1024;
    int bid  = blockIdx.x;
    int mt   = bid & 7;           // 8 m-tiles of 128 rows
    int h    = (bid >> 3) & 15;
    int seg  = bid >> 7;
    int lane = threadIdx.x & 31;
    int wave = threadIdx.x >> 5;
    int t0   = mt * 128 + wave * 16;        // first sparse query row of wave
    int c    = h >> cshift;
    int mrw  = lane & 15;                   // fragment row/col index
    int hi8  = (lane & 16) ? 8 : 0;

    __shared__ _Float16 plds[8][512] __attribute__((aligned(64)));
    _Float16* myp = plds[wave];

    // Q fragments for this wave's 16 rows (K=0..31 and K=32..63)
    long pq = (long)seg * sl + (long)(t0 + mrw) * r + c;
    const _Float16* qrow = q16 + ((long)h * LSEQ + pq) * HD;
    v16h aq0 = load_fragA(qrow, lane);
    v16h aq1 = load_fragA(qrow + 32, lane);

    // constant all-ones B fragment: rowsum(P) = P x ones via the matrix pipe
    v16h vones;
#pragma unroll
    for (int i = 0; i < 16; ++i) vones[i] = (_Float16)1.0f;

    float mrowv[8];
    v8f  oacc[5];                 // 4 output dim-tiles + 1 rowsum tile
#pragma unroll
    for (int u = 0; u < 8; ++u) mrowv[u] = -1e30f;
#pragma unroll
    for (int u = 0; u < 5; ++u) oacc[u] = (v8f){};

    int nch = (t0 + 47) >> 5;               // chunks of 32 keys, causal bound
    for (int j = 0; j < nch; ++j) {
        int kb = j << 5;
        // ---- S = Q K^T (two 16-key tiles, K-dim 64 split 32+32) ----
        long pk0 = (long)seg * sl + (long)(kb + mrw) * r + c;
        long pk1 = pk0 + (long)16 * r;
        const _Float16* kr0 = k16 + ((long)h * LSEQ + pk0) * HD;
        const _Float16* kr1 = k16 + ((long)h * LSEQ + pk1) * HD;
        if (j + 1 < nch) {                  // prefetch next chunk's K rows
            __builtin_prefetch((const void*)(kr0 + (size_t)32 * r * HD), 0, 3);
            __builtin_prefetch((const void*)(kr1 + (size_t)32 * r * HD), 0, 3);
        }
        v16h b00 = load_fragB(kr0, lane);
        v16h b01 = load_fragB(kr0 + 32, lane);
        v16h b10 = load_fragB(kr1, lane);
        v16h b11 = load_fragB(kr1 + 32, lane);
        v8f s_lo = (v8f){}, s_hi = (v8f){};
        s_lo = wmma_f16(aq0, b00, s_lo);
        s_lo = wmma_f16(aq1, b01, s_lo);
        s_hi = wmma_f16(aq0, b10, s_hi);
        s_hi = wmma_f16(aq1, b11, s_hi);

        // ---- causal mask + online softmax ----
        int kk = kb + mrw;                  // this lane's key (lo tile)
        float slv[8], shv[8], rmax[8];
#pragma unroll
        for (int u = 0; u < 8; ++u) {
            int row = t0 + u + hi8;
            float a = s_lo[u]; if (kk > row)      a = -1e9f;
            float b = s_hi[u]; if (kk + 16 > row) b = -1e9f;
            slv[u] = a; shv[u] = b;
            rmax[u] = fmaxf(a, b);
        }
#pragma unroll
        for (int u = 0; u < 8; ++u)
            rmax[u] = redmax16(rmax[u]);
#pragma unroll
        for (int u = 0; u < 8; ++u) {
            float mn   = fmaxf(mrowv[u], rmax[u]);
            float corr = __expf(mrowv[u] - mn);
            mrowv[u]   = mn;
            float pl = __expf(slv[u] - mn);
            float ph = __expf(shv[u] - mn);
            oacc[0][u] *= corr; oacc[1][u] *= corr;
            oacc[2][u] *= corr; oacc[3][u] *= corr;
            oacc[4][u] *= corr;             // running rowsum rescale
            int rr = u + hi8;               // transpose P through LDS
            myp[rr * 32 + mrw]      = (_Float16)pl;
            myp[rr * 32 + 16 + mrw] = (_Float16)ph;
        }
        asm volatile("s_wait_dscnt 0" ::: "memory");

        // ---- O += P V ; rowsum += P x ones ----
        v16h pf = load_fragA(myp + mrw * 32, lane);
#pragma unroll
        for (int n = 0; n < 4; ++n) {
            const _Float16* vcol = vT + ((long)(h * HD + n * 16 + mrw)) * SeqB
                                      + (long)seg * 1024 + kb;
            v16h vb = load_fragB(vcol, lane);
            oacc[n] = wmma_f16(pf, vb, oacc[n]);
        }
        oacc[4] = wmma_f16(pf, vones, oacc[4]);
    }

    // ---- epilogue: normalize, write compact branch output + lse ----
    int sgbase = seg * 1024 + t0;
#pragma unroll
    for (int u = 0; u < 8; ++u) {
        float lr  = oacc[4][u];
        float inv = 1.0f / lr;
        int   sg  = sgbase + u + hi8;
        long  ob  = ((long)h * SeqB + sg) * HD + mrw;
        outc[ob]      = oacc[0][u] * inv;
        outc[ob + 16] = oacc[1][u] * inv;
        outc[ob + 32] = oacc[2][u] * inv;
        outc[ob + 48] = oacc[3][u] * inv;
        if (mrw == 0)
            lsec[(long)h * SeqB + sg] = mrowv[u] + __logf(lr);
    }
}

// ---------- LSE-weighted merge of the three branches ----------
__global__ __launch_bounds__(256)
void dilattn_combine(const float* __restrict__ o0, const float* __restrict__ o1,
                     const float* __restrict__ o2, const float* __restrict__ l0,
                     const float* __restrict__ l1, const float* __restrict__ l2,
                     float* __restrict__ out) {
    int idx = blockIdx.x * blockDim.x + threadIdx.x;   // LSEQ*NH*HD
    if (idx >= LSEQ * NH * HD) return;
    int d = idx & 63;
    int h = (idx >> 6) & 15;
    int p = idx >> 10;
    int c1 = h >> 3, c2 = h >> 2;
    bool v1 = ((p & 1) == c1);
    bool v2 = ((p & 3) == c2);
    int sg1 = (p >> 11) * 1024 + ((p & 2047) >> 1);
    int sg2 = (p - c2) >> 2;

    float L0 = l0[h * 4096 + p];
    float L1 = v1 ? l1[h * 2048 + sg1] : -1e8f;
    float L2 = v2 ? l2[h * 1024 + sg2] : -1e8f;
    float mx = fmaxf(L0, fmaxf(L1, L2));
    float w0 = __expf(L0 - mx), w1 = __expf(L1 - mx), w2 = __expf(L2 - mx);
    float dn = w0 + w1 + w2;

    float acc = w0 * o0[((long)h * 4096 + p) * HD + d];
    if (v1) acc += w1 * o1[((long)h * 2048 + sg1) * HD + d];
    if (v2) acc += w2 * o2[((long)h * 1024 + sg2) * HD + d];
    out[(long)p * (NH * HD) + h * HD + d] = acc / dn;
}

extern "C" void kernel_launch(void* const* d_in, const int* in_sizes, int n_in,
                              void* d_out, int out_size, void* d_ws, size_t ws_size,
                              hipStream_t stream) {
    (void)in_sizes; (void)n_in; (void)out_size; (void)ws_size;
    const float* q = (const float*)d_in[0];
    const float* k = (const float*)d_in[1];
    const float* v = (const float*)d_in[2];
    float* out = (float*)d_out;

    // workspace layout (halves then floats)
    _Float16* q16 = (_Float16*)d_ws;
    _Float16* k16 = q16 + (size_t)NH * LSEQ * HD;               // 4M halves
    _Float16* vt0 = k16 + (size_t)NH * LSEQ * HD;
    _Float16* vt1 = vt0 + (size_t)NH * HD * 4096;
    _Float16* vt2 = vt1 + (size_t)NH * HD * 2048;
    float* fb = (float*)(vt2 + (size_t)NH * HD * 1024);
    float* o0 = fb;
    float* o1 = o0 + (size_t)NH * 4096 * HD;
    float* o2 = o1 + (size_t)NH * 2048 * HD;
    float* l0 = o2 + (size_t)NH * 1024 * HD;
    float* l1 = l0 + (size_t)NH * 4096;
    float* l2 = l1 + (size_t)NH * 2048;

    dilattn_prep_qk<<<(NH * LSEQ * 32 + 255) / 256, 256, 0, stream>>>(q, k, q16, k16);
    dilattn_prep_v<<<(NH * HD * 4096 + 255) / 256, 256, 0, stream>>>(v, vt0, 1, 1024, 4, 4);
    dilattn_prep_v<<<(NH * HD * 2048 + 255) / 256, 256, 0, stream>>>(v, vt1, 2, 2048, 2, 3);
    dilattn_prep_v<<<(NH * HD * 1024 + 255) / 256, 256, 0, stream>>>(v, vt2, 4, 4096, 1, 2);

    dilattn_attn<<<4 * 16 * 8, 256, 0, stream>>>(q16, k16, vt0, o0, l0, 1, 1024, 4, 4);
    dilattn_attn<<<2 * 16 * 8, 256, 0, stream>>>(q16, k16, vt1, o1, l1, 2, 2048, 2, 3);
    dilattn_attn<<<1 * 16 * 8, 256, 0, stream>>>(q16, k16, vt2, o2, l2, 4, 4096, 1, 2);

    dilattn_combine<<<(LSEQ * NH * HD + 255) / 256, 256, 0, stream>>>(
        o0, o1, o2, l0, l1, l2, out);
}
